// RelPosEncoder_26534307955239
// MI455X (gfx1250) — matrix operand order repaired and checked by reading, not compiled, not run on hardware
//
#include <hip/hip_runtime.h>
#include <hip/hip_bf16.h>

typedef __attribute__((ext_vector_type(16))) _Float16 v16h;
typedef __attribute__((ext_vector_type(8)))  float    v8f;
typedef __attribute__((ext_vector_type(4)))  float    v4f;

#define Bdim 4
#define Mdim 512
#define Ndim 512
#define NT   (Ndim / 16)   // 32 n-tiles
#define MB   (Mdim / 64)   // 8 m-blocks (64 m per block, 16 per wave)

// ---------------------------------------------------------------------------
// Kernel 1: per-row MLP front-end.
//   Aproj[b,m,k] = relu(trk_in24 . W1 + b1) . W2[:64]  + b2[k]   (b2 folded)
//   Bproj[b,n,k] = relu(det_in24 . W1 + b1) . W2[64:]            (no bias)
// One thread per row; the 64-d hidden vector lives in LDS so loops stay rolled.
// ---------------------------------------------------------------------------
__global__ __launch_bounds__(128)
void rpe_encode_project(const float* __restrict__ trk_rel,
                        const float* __restrict__ det_rel,
                        const float* __restrict__ trk_bbox,
                        const float* __restrict__ det_bbox,
                        const float* __restrict__ W1,
                        const float* __restrict__ b1,
                        const float* __restrict__ W2,
                        const float* __restrict__ b2,
                        float* __restrict__ Ap,
                        float* __restrict__ Bp) {
    __shared__ float smem[128 * 64];
    const int tid = threadIdx.x;
    const int row = blockIdx.x * 128 + tid;          // 0 .. 4095
    const int BM  = Bdim * Mdim;
    const bool is_trk = row < BM;
    const int  r = is_trk ? row : row - BM;

    const float* bbox = (is_trk ? trk_bbox : det_bbox) + (size_t)r * 4;
    const float* rel  = (is_trk ? trk_rel  : det_rel ) + (size_t)r * 20;

    float in[24];
#pragma unroll
    for (int i = 0; i < 4; ++i)  in[i]     = bbox[i];
#pragma unroll
    for (int i = 0; i < 20; ++i) in[4 + i] = rel[i];

    float* e = &smem[tid * 64];
    for (int j = 0; j < 64; ++j) {
        float s = b1[j];
#pragma unroll
        for (int i = 0; i < 24; ++i) s += in[i] * W1[i * 64 + j];
        e[j] = fmaxf(s, 0.0f);
    }
    __syncthreads();  // (waves only read their own rows; cheap safety)

    const int   woff = is_trk ? 0 : 64;
    float* outp = (is_trk ? Ap : Bp) + (size_t)r * 64;
    for (int k = 0; k < 64; ++k) {
        float s = is_trk ? b2[k] : 0.0f;
        for (int j = 0; j < 64; ++j) s += e[j] * W2[(size_t)(woff + j) * 64 + k];
        outp[k] = s;
    }
}

// ---------------------------------------------------------------------------
// Kernel 2: pairwise head via v_wmma_f32_16x16x32_f16.
//   For a 16-wide n-tile and one m:  H(16x64) = relu(Ap[m] + Bp[ntile] )
//   D(16x16) = H x W3 (+ b3 via C init), K=64 as two chained K=32 WMMAs.
// A-fragment layout (16-bit A 16x32): lane hi=lane>>4, row M=lane&15,
//   element j holds K = c*32 + hi*8 + ((j&8)?16:0) + (j&7)  -> contiguous
//   float4 groups at offsets {0,4,16,20} per (c,hi).
// B-fragment (W3, 32x16): N = lane&15, K = c*32 + hi*8 + ((j&8)?16:0) + (j&7), same as A.
// ---------------------------------------------------------------------------
__global__ __launch_bounds__(128)
void rpe_pair_head(const float* __restrict__ Ap,
                   const float* __restrict__ Bp,
                   const float* __restrict__ W3,
                   const float* __restrict__ b3,
                   float* __restrict__ out) {
    const int wave = threadIdx.x >> 5;
    const int lane = threadIdx.x & 31;
    const int hi   = (lane >> 4) & 1;
    const int col  = lane & 15;

    int blk = blockIdx.x;                 // [0, Bdim*NT*MB)
    const int b   = blk / (NT * MB);
    int rem       = blk % (NT * MB);
    const int nt  = rem / MB;
    const int mb  = rem % MB;
    const int m0  = mb * 64 + wave * 16;  // 16 m's per wave
    const int n   = nt * 16 + col;        // this lane's n-row in the A fragment

    // W3 fragments (f16), loaded once per wave.
    v16h w3f0, w3f1;
#pragma unroll
    for (int j = 0; j < 16; ++j) {
        w3f0[j] = (_Float16)W3[(hi * 8 + (j & 7) + (j >> 3) * 16) * 16 + col];
        w3f1[j] = (_Float16)W3[(32 + hi * 8 + (j & 7) + (j >> 3) * 16) * 16 + col];
    }
    const float b3c = b3[col];

    // B-side (det) fragment values for this n, loaded once per wave.
    const int ofs[4] = {0, 4, 16, 20};
    const float* bq = Bp + ((size_t)b * Ndim + n) * 64 + hi * 8;
    v4f bv[2][4];
#pragma unroll
    for (int c = 0; c < 2; ++c)
#pragma unroll
        for (int g = 0; g < 4; ++g)
            bv[c][g] = *(const v4f*)(bq + c * 32 + ofs[g]);

    const float* aq0 = Ap + ((size_t)b * Mdim + m0) * 64 + hi * 8;

    for (int mi = 0; mi < 16; ++mi) {
        const float* aq = aq0 + mi * 64;
        v4f av[2][4];
#pragma unroll
        for (int c = 0; c < 2; ++c)
#pragma unroll
            for (int g = 0; g < 4; ++g)
                av[c][g] = *(const v4f*)(aq + c * 32 + ofs[g]);

        // h = relu(A + B) in f16 fragments.
        v16h a0, a1;
#pragma unroll
        for (int g = 0; g < 4; ++g)
#pragma unroll
            for (int e = 0; e < 4; ++e) {
                a0[g * 4 + e] = (_Float16)fmaxf(av[0][g][e] + bv[0][g][e], 0.0f);
                a1[g * 4 + e] = (_Float16)fmaxf(av[1][g][e] + bv[1][g][e], 0.0f);
            }

        v8f acc;
#pragma unroll
        for (int rr = 0; rr < 8; ++rr) acc[rr] = b3c;   // fold b3 into C

        acc = __builtin_amdgcn_wmma_f32_16x16x32_f16(false, a0, false, w3f0,
                                                     (short)0, acc, false, false);
        acc = __builtin_amdgcn_wmma_f32_16x16x32_f16(false, a1, false, w3f1,
                                                     (short)0, acc, false, false);

        // D layout: row M = rr + hi*8 (n within tile), col = o. Streamed NT stores.
        float* op = out + (((size_t)b * Mdim + (m0 + mi)) * Ndim + nt * 16 + hi * 8) * 16 + col;
#pragma unroll
        for (int rr = 0; rr < 8; ++rr)
            __builtin_nontemporal_store(acc[rr], op + (size_t)rr * 16);
    }
}

// ---------------------------------------------------------------------------
extern "C" void kernel_launch(void* const* d_in, const int* in_sizes, int n_in,
                              void* d_out, int out_size, void* d_ws, size_t ws_size,
                              hipStream_t stream) {
    const float* trk_rel  = (const float*)d_in[0];
    const float* det_rel  = (const float*)d_in[1];
    const float* trk_bbox = (const float*)d_in[2];
    const float* det_bbox = (const float*)d_in[3];
    const float* W1       = (const float*)d_in[4];
    const float* b1       = (const float*)d_in[5];
    const float* W2       = (const float*)d_in[6];
    const float* b2       = (const float*)d_in[7];
    const float* W3       = (const float*)d_in[8];
    const float* b3       = (const float*)d_in[9];
    float* out = (float*)d_out;

    float* Ap = (float*)d_ws;                       // Bdim*Mdim*64 floats (512 KB)
    float* Bp = Ap + (size_t)Bdim * Mdim * 64;      // Bdim*Ndim*64 floats (512 KB)

    // Stage 1: row embeddings + projections (4096 rows / 128 threads = 32 blocks)
    rpe_encode_project<<<(Bdim * (Mdim + Ndim)) / 128, 128, 0, stream>>>(
        trk_rel, det_rel, trk_bbox, det_bbox, W1, b1, W2, b2, Ap, Bp);

    // Stage 2: pairwise WMMA head. 4*32*8 = 1024 blocks x 4 waves.
    rpe_pair_head<<<Bdim * NT * MB, 128, 0, stream>>>(Ap, Bp, W3, b3, out);
}
